// WindowAttention_60344290509101
// MI455X (gfx1250) — matrix operand order repaired and checked by reading, not compiled, not run on hardware
//
#include <hip/hip_runtime.h>
#include <hip/hip_bf16.h>
#include <math.h>
#include <stdint.h>

typedef __attribute__((ext_vector_type(16))) __bf16 v16bf;
typedef __attribute__((ext_vector_type(8)))  float  v8f;

namespace {
constexpr int kHeads = 8;
constexpr int kWinQ  = 64;
constexpr int kTopK  = 64;
constexpr int kKtot  = 128;   // 2*TOPK
constexpr int kC     = 512;
constexpr int kNW    = 128;
constexpr int kB     = 8;
constexpr int kLq    = 8192;
constexpr int kLk0   = 8192;
constexpr int kLk1   = 4096;
}

// ---------------------------------------------------------------- WMMA helpers

__device__ __forceinline__ v8f wmma_bf16(v16bf a, v16bf b, v8f c) {
#if defined(__HIP_DEVICE_COMPILE__)
  return __builtin_amdgcn_wmma_f32_16x16x32_bf16(false, a, false, b, (short)0, c,
                                                 false, false);
#else
  (void)a; (void)b;
  return c;
#endif
}

// A operand (16x32, M x K): lane L (0-15) = row L, k in {0..7,16..23};
// lane 16+L = row L, k in {8..15,24..31}. Two 16B LDS loads.
__device__ __forceinline__ v16bf frag_ld_a(const __bf16* lds, int strideElems,
                                           int row, int kElem, int lane) {
  union { v16bf v; uint4 q[2]; } f;
  const char* p = (const char*)(lds + row * strideElems + kElem) + ((lane >> 4) << 4);
  f.q[0] = *(const uint4*)(p);
  f.q[1] = *(const uint4*)(p + 32);
  return f.v;
}

// B operand (32x16, K x N), stored transposed in LDS as [n][k]:
// lane L (0-15) = col L, k in {0..7,16..23}; lane 16+L = col L, k in {8..15,24..31}.
__device__ __forceinline__ v16bf frag_ld_b(const __bf16* lds, int strideElems,
                                           int col, int kElem, int lane) {
  union { v16bf v; uint4 q[2]; } f;
  const char* p = (const char*)(lds + col * strideElems + kElem) + ((lane >> 4) << 4);
  f.q[0] = *(const uint4*)(p);
  f.q[1] = *(const uint4*)(p + 32);
  return f.v;
}

// ------------------------------------------- CDNA5 async global->LDS (16B/lane)
// GLOBAL_LOAD_ASYNC_TO_LDS_B128 (GV mode): vdst = wave-relative LDS byte addr
// (low 32 bits of a generic shared pointer), vaddr = 64-bit global address.
// Tracked by ASYNCcnt; retire with s_wait_asynccnt before the block barrier.
__device__ __forceinline__ void async_copy_b128(void* ldsDst, const void* gsrc) {
#if defined(__HIP_DEVICE_COMPILE__)
  unsigned lds = (unsigned)(uintptr_t)ldsDst;
  asm volatile("global_load_async_to_lds_b128 %0, %1, off"
               :
               : "v"(lds), "v"(gsrc)
               : "memory");
#else
  (void)ldsDst; (void)gsrc;
#endif
}

__device__ __forceinline__ void async_wait0() {
#if defined(__HIP_DEVICE_COMPILE__)
  asm volatile("s_wait_asynccnt 0x0" ::: "memory");
#endif
}

// ------------------------------------------------------- top-k closest indices
// Reproduces np.argsort(|cq-j|, stable)[:, :64]: distance-major, lower j first.
__global__ void build_idx_kernel(int* __restrict__ idx0, int* __restrict__ idx1) {
  int t = threadIdx.x;
  if (t >= 256) return;
  int lvl = t >> 7, iw = t & 127;
  int L = lvl ? kLk1 : kLk0;
  int f = L / kNW;
  int cq = iw * f + f / 2;
  int* out = (lvl ? idx1 : idx0) + iw * kTopK;
  int cnt = 0;
  for (int d = 0; cnt < kTopK; ++d) {
    int jm = cq - d;
    if (jm >= 0 && jm < L) {
      out[cnt++] = jm;
      if (cnt == kTopK) break;
    }
    if (d > 0) {
      int jp = cq + d;
      if (jp < L) out[cnt++] = jp;
    }
  }
}

// ------------------------------------------------------------ WMMA GEMM + bias
// Y[M,N] = X[M,K] @ W[K,N] + bias ; block tile 128x128, 8 waves (2x4),
// wave tile 64x32 (4x2 of 16x16), K-step 32 (one wmma per tile per step).
template <typename TIn, bool OUT_BF16>
__global__ __launch_bounds__(256) void gemm_bias_wmma(
    const TIn* __restrict__ X, const float* __restrict__ W,
    const float* __restrict__ bias, void* __restrict__ Yv,
    int M, int N, int K) {
  __shared__ __bf16 ldsA[128 * 40];  // [row][k], stride 40 elems (80B, 16B-aligned)
  __shared__ __bf16 ldsB[128 * 40];  // [n][k] transposed

  const int tid = threadIdx.x;
  const int lane = tid & 31, wid = tid >> 5;
  const int l16 = lane & 15, hi8 = (lane >> 4) << 3;
  const int wM = wid >> 2, wN = wid & 3;
  const size_t row0 = (size_t)blockIdx.y * 128;
  const int col0 = blockIdx.x * 128;

  const v8f zero = {0.f, 0.f, 0.f, 0.f, 0.f, 0.f, 0.f, 0.f};
  v8f acc[4][2];
#pragma unroll
  for (int m = 0; m < 4; ++m)
#pragma unroll
    for (int n = 0; n < 2; ++n) acc[m][n] = zero;

  for (int k0 = 0; k0 < K; k0 += 32) {
#pragma unroll
    for (int i = 0; i < 16; ++i) {          // A: 128x32, coalesced on k
      int id = tid + 256 * i;
      int r = id >> 5, c = id & 31;
      float v = (float)X[(row0 + r) * K + (k0 + c)];
      ldsA[r * 40 + c] = (__bf16)v;
    }
#pragma unroll
    for (int i = 0; i < 16; ++i) {          // B: 32x128, coalesced on n, transpose
      int id = tid + 256 * i;
      int kk = id >> 7, n = id & 127;
      float v = W[(size_t)(k0 + kk) * N + (col0 + n)];
      ldsB[n * 40 + kk] = (__bf16)v;
    }
    __syncthreads();
    v16bf afr[4];
#pragma unroll
    for (int mt = 0; mt < 4; ++mt)
      afr[mt] = frag_ld_a(ldsA, 40, wM * 64 + mt * 16 + l16, 0, lane);
#pragma unroll
    for (int nt = 0; nt < 2; ++nt) {
      v16bf bfr = frag_ld_b(ldsB, 40, wN * 32 + nt * 16 + l16, 0, lane);
#pragma unroll
      for (int mt = 0; mt < 4; ++mt)
        acc[mt][nt] = wmma_bf16(afr[mt], bfr, acc[mt][nt]);
    }
    __syncthreads();
  }

#pragma unroll
  for (int mt = 0; mt < 4; ++mt)
#pragma unroll
    for (int nt = 0; nt < 2; ++nt) {
      int col = col0 + wN * 32 + nt * 16 + l16;
      float bv = bias[col];
#pragma unroll
      for (int r = 0; r < 8; ++r) {
        size_t row = row0 + wM * 64 + mt * 16 + hi8 + r;
        float v = acc[mt][nt][r] + bv;
        if (OUT_BF16)
          ((__bf16*)Yv)[row * (size_t)N + col] = (__bf16)v;
        else
          ((float*)Yv)[row * (size_t)N + col] = v;
      }
    }
}

// --------------------------------------------------- fused windowed attention
// One block per (window, head): S = scale*QK^T (+mask), softmax, O = P V.
#define SM_Q_OFF 0          // Q  bf16 [64][72]   = 9216 B   (dead after QK)
#define SM_K_OFF 9216       // K  bf16 [128][72]  = 18432 B  (dead after QK)
#define SM_VT_OFF 34048     // V^T bf16 [64][136] = 17408 B
#define SM_P_OFF 51456      // P  bf16 [64][136]  = 17408 B
#define SM_MV_OFF 68864     // mask add f32 [128] = 512 B
#define SM_CNT_OFF 69376    // int
#define SM_BYTES 69632      // S f32 [64][132]=33792B overlays Q+K region

__global__ __launch_bounds__(256) void attn_wmma_kernel(
    const __bf16* __restrict__ Qbf, const __bf16* __restrict__ KV0,
    const __bf16* __restrict__ KV1, const int* __restrict__ idx0,
    const int* __restrict__ idx1, const unsigned char* __restrict__ mask0,
    const unsigned char* __restrict__ mask1, float* __restrict__ attnOut,
    __bf16* __restrict__ Obf) {
  extern __shared__ char smem[];
  __bf16* ldsQ  = (__bf16*)(smem + SM_Q_OFF);
  __bf16* ldsK  = (__bf16*)(smem + SM_K_OFF);
  float*  ldsS  = (float*)(smem);            // overlays Q/K after barrier
  __bf16* ldsVT = (__bf16*)(smem + SM_VT_OFF);
  __bf16* ldsP  = (__bf16*)(smem + SM_P_OFF);
  float*  mval  = (float*)(smem + SM_MV_OFF);
  int*    cntp  = (int*)(smem + SM_CNT_OFF);

  const int tid = threadIdx.x;
  const int lane = tid & 31, wid = tid >> 5;
  const int l16 = lane & 15, hi8 = (lane >> 4) << 3;
  const int h = blockIdx.x;
  const int w = blockIdx.y;
  const int b = w >> 7, iw = w & 127;

  if (tid == 0) *cntp = 0;
  __syncthreads();

  // Q tile 64x64 (head slice): async global->LDS, 16B per lane, no VGPR staging
#pragma unroll
  for (int i = 0; i < 2; ++i) {
    int id = tid + 256 * i;
    int qr = id >> 3, q4 = id & 7;
    async_copy_b128(ldsQ + qr * 72 + q4 * 8,
                    Qbf + ((size_t)(b * kLq + iw * 64 + qr)) * kC + h * 64 + q4 * 8);
  }

  // gathered mask -> additive -inf terms + all-masked count
  if (tid < 128) {
    int lvl = tid >> 6, tt = tid & 63;
    int j = lvl ? idx1[iw * 64 + tt] : idx0[iw * 64 + tt];
    unsigned char m = lvl ? mask1[b * kLk1 + j] : mask0[b * kLk0 + j];
    mval[tid] = m ? -INFINITY : 0.0f;
    if (m) atomicAdd(cntp, 1);
  }

  // gather K (async copy, row-major) and V (transposed [d][key]) for 128 keys
#pragma unroll
  for (int i = 0; i < 4; ++i) {
    int id = tid + 256 * i;
    int row = id >> 3, c4 = id & 7;
    int lvl = row >> 6, tt = row & 63;
    const __bf16* src;
    if (!lvl) {
      int j = idx0[iw * 64 + tt];
      src = KV0 + ((size_t)(b * kLk0 + j)) * (2 * kC) + h * 64 + c4 * 8;
    } else {
      int j = idx1[iw * 64 + tt];
      src = KV1 + ((size_t)(b * kLk1 + j)) * (2 * kC) + h * 64 + c4 * 8;
    }
    async_copy_b128(ldsK + row * 72 + c4 * 8, src);            // k half
    union { uint4 q; __bf16 e[8]; } u;
    u.q = *(const uint4*)(src + kC);                           // v half
#pragma unroll
    for (int j2 = 0; j2 < 8; ++j2)
      ldsVT[(c4 * 8 + j2) * 136 + row] = u.e[j2];
  }
  async_wait0();     // retire this wave's async copies before the barrier
  __syncthreads();

  // S = scale * Q K^T + mask   (64x128, 4x2 wave grid of 16x64 tiles)
  {
    const int wM = wid >> 1, wN = wid & 1;
    const v8f zero = {0.f, 0.f, 0.f, 0.f, 0.f, 0.f, 0.f, 0.f};
    v8f sacc[4] = {zero, zero, zero, zero};
#pragma unroll
    for (int ks = 0; ks < 2; ++ks) {
      v16bf a = frag_ld_a(ldsQ, 72, wM * 16 + l16, ks * 32, lane);
#pragma unroll
      for (int nt = 0; nt < 4; ++nt) {
        v16bf bb = frag_ld_b(ldsK, 72, wN * 64 + nt * 16 + l16, ks * 32, lane);
        sacc[nt] = wmma_bf16(a, bb, sacc[nt]);
      }
    }
    __syncthreads();  // Q/K dead; mask count final -> safe to overlay with S
    const bool allm = (*cntp == 128);
    const int rb = wM * 16 + hi8;
#pragma unroll
    for (int nt = 0; nt < 4; ++nt) {
      int col = wN * 64 + nt * 16 + l16;
      float ma = mval[col];
#pragma unroll
      for (int r = 0; r < 8; ++r) {
        float v = sacc[nt][r] * 0.125f + ma;  // scale = hd^-0.5 = 1/8
        if (allm) v = 0.0f;                   // all-masked row -> zeros
        ldsS[(rb + r) * 132 + col] = v;
      }
    }
  }
  __syncthreads();

  // fp32 softmax per row; emit attn (fp32) and P (bf16)
  if (tid < 64) {
    const float* Sr = ldsS + tid * 132;
    float mx = -INFINITY;
    for (int k = 0; k < kKtot; ++k) mx = fmaxf(mx, Sr[k]);
    float sum = 0.0f;
    for (int k = 0; k < kKtot; ++k) sum += expf(Sr[k] - mx);
    float inv = 1.0f / sum;
    float* ao = attnOut + (((size_t)w * kHeads + h) * kWinQ + tid) * kKtot;
    for (int k = 0; k < kKtot; ++k) {
      float p = expf(Sr[k] - mx) * inv;
      ao[k] = p;
      ldsP[tid * 136 + k] = (__bf16)p;
    }
  }
  __syncthreads();

  // O = P V   (64x64, 4x2 wave grid of 16x32 tiles, K=128 in 4 steps)
  {
    const int wM = wid >> 1, wN = wid & 1;
    const v8f zero = {0.f, 0.f, 0.f, 0.f, 0.f, 0.f, 0.f, 0.f};
    v8f oacc[2] = {zero, zero};
#pragma unroll
    for (int ks = 0; ks < 4; ++ks) {
      v16bf a = frag_ld_a(ldsP, 136, wM * 16 + l16, ks * 32, lane);
#pragma unroll
      for (int nt = 0; nt < 2; ++nt) {
        v16bf bb = frag_ld_b(ldsVT, 136, wN * 32 + nt * 16 + l16, ks * 32, lane);
        oacc[nt] = wmma_bf16(a, bb, oacc[nt]);
      }
    }
    const int rb = wM * 16 + hi8;
#pragma unroll
    for (int nt = 0; nt < 2; ++nt) {
      int d = wN * 32 + nt * 16 + l16;
#pragma unroll
      for (int r = 0; r < 8; ++r) {
        int m = rb + r;
        Obf[((size_t)(b * kLq + iw * 64 + m)) * kC + h * 64 + d] =
            (__bf16)oacc[nt][r];
      }
    }
  }
}

// ---------------------------------------------------------------------- launch
extern "C" void kernel_launch(void* const* d_in, const int* in_sizes, int n_in,
                              void* d_out, int out_size, void* d_ws, size_t ws_size,
                              hipStream_t stream) {
  (void)in_sizes; (void)n_in; (void)out_size; (void)ws_size;
  const float* query = (const float*)d_in[0];
  const float* key0  = (const float*)d_in[1];
  const float* key1  = (const float*)d_in[2];
  const unsigned char* mask0 = (const unsigned char*)d_in[3];
  const unsigned char* mask1 = (const unsigned char*)d_in[4];
  const float* Wq  = (const float*)d_in[5];
  const float* bq  = (const float*)d_in[6];
  const float* Wkv = (const float*)d_in[7];
  const float* bkv = (const float*)d_in[8];
  const float* Wpr = (const float*)d_in[9];
  const float* bpr = (const float*)d_in[10];

  char* ws = (char*)d_ws;
  int* idx0 = (int*)ws;                                  // 128*64*4
  int* idx1 = (int*)(ws + 32768);                        // 128*64*4
  __bf16* Qbf = (__bf16*)(ws + 65536);                   // 65536 x 512
  __bf16* KV0 = Qbf + (size_t)kB * kLq * kC;             // 65536 x 1024
  __bf16* KV1 = KV0 + (size_t)kB * kLk0 * 2 * kC;        // 32768 x 1024
  __bf16* Obf = KV1 + (size_t)kB * kLk1 * 2 * kC;        // 65536 x 512

  float* outMat  = (float*)d_out;
  float* attnOut = outMat + (size_t)kB * kLq * kC;

  build_idx_kernel<<<1, 256, 0, stream>>>(idx0, idx1);

  gemm_bias_wmma<float, true>
      <<<dim3(kC / 128, (kB * kLq) / 128), 256, 0, stream>>>(
          query, Wq, bq, (void*)Qbf, kB * kLq, kC, kC);
  gemm_bias_wmma<float, true>
      <<<dim3((2 * kC) / 128, (kB * kLk0) / 128), 256, 0, stream>>>(
          key0, Wkv, bkv, (void*)KV0, kB * kLk0, 2 * kC, kC);
  gemm_bias_wmma<float, true>
      <<<dim3((2 * kC) / 128, (kB * kLk1) / 128), 256, 0, stream>>>(
          key1, Wkv, bkv, (void*)KV1, kB * kLk1, 2 * kC, kC);

  attn_wmma_kernel<<<dim3(kHeads, kB * kNW), 256, SM_BYTES, stream>>>(
      Qbf, KV0, KV1, idx0, idx1, mask0, mask1, attnOut, Obf);

  gemm_bias_wmma<__bf16, false>
      <<<dim3(kC / 128, (kB * kLq) / 128), 256, 0, stream>>>(
          Obf, Wpr, bpr, d_out, kB * kLq, kC, kC);
}
